// RSNN_1666447311400
// MI455X (gfx1250) — compile-verified, matrix operand order checked by reading, not run on hardware
//
#include <hip/hip_runtime.h>
#include <hip/hip_bf16.h>

// ---------------- problem constants (match reference) ----------------
#define T_STEPS 1024
#define NB      16      // batch
#define NOBS    1024    // observed neurons
#define NN      2048    // total neurons
#define DECAY   0.5f
#define THRESH  1.0f

typedef __attribute__((ext_vector_type(16))) _Float16 v16h;
typedef __attribute__((ext_vector_type(8)))  _Float16 v8h;
typedef __attribute__((ext_vector_type(8)))  float    v8f;

// workspace layout
#define W16_BYTES   ((size_t)NN * NN * 2)            // 8 MB  f16 copy of W
#define SBUF_BYTES  ((size_t)2 * NB * NN * 2)        // 128 KB double-buffered spikes (f16)
// barrier counter lives right after

#define NUM_WGS       64
#define WAVES_PER_WG  2   // 64*2 = 128 waves = 128 tiles of 16 output columns

// ---------------------------------------------------------------------
// Prep: W (f32, row-major [n][k]) -> f16 copy; zero spike buffers + barrier.
// Run every launch so graph replays are deterministic.
__global__ void rsnn_prep(const float* __restrict__ W,
                          _Float16* __restrict__ Wf16,
                          _Float16* __restrict__ sbuf,
                          unsigned* __restrict__ ctr) {
    size_t i = (size_t)blockIdx.x * blockDim.x + threadIdx.x;
    size_t stride = (size_t)gridDim.x * blockDim.x;
    for (size_t j = i; j < (size_t)NN * NN; j += stride)
        Wf16[j] = (_Float16)W[j];
    for (size_t j = i; j < (size_t)2 * NB * NN; j += stride)
        sbuf[j] = (_Float16)0.0f;
    if (i == 0) *ctr = 0u;
}

// ---------------------------------------------------------------------
// Persistent RSNN kernel. One wave owns a 16x16 C tile (batch x out-neuron);
// membrane potential u lives in the WMMA f32 accumulator for all 1024 steps.
__global__ void rsnn_step_kernel(const float* __restrict__ spike_input, // [T,B,NOBS] f32
                                 const float* __restrict__ input_weight, // [NOBS] f32
                                 const _Float16* __restrict__ Wf16,      // [NN,NN] row-major
                                 _Float16* __restrict__ sbuf,            // [2,NB,NN] f16
                                 unsigned* __restrict__ ctr,
                                 float* __restrict__ out)                // [T,B,NOBS] f32
{
    const int wave   = threadIdx.x >> 5;
    const int lane   = threadIdx.x & 31;
    const int tile   = blockIdx.x * WAVES_PER_WG + wave;  // 0..127
    const int n_base = tile * 16;
    const int h      = lane >> 4;   // K-half select (ISA f16 fragment layout)
    const int mn     = lane & 15;   // A: row m   B/C: col n

    const int  n_global = n_base + mn;
    const bool observed = (n_base < NOBS);   // uniform per wave
    const float wscale  = observed ? input_weight[n_global] : 0.0f;

    // B operand column n of W^T == row n of W (row-major, contiguous in K)
    const _Float16* __restrict__ wrow = Wf16 + (size_t)n_global * NN;

    v8f acc = {};  // u = 0 initial membrane potential

    for (int t = 0; t < T_STEPS; ++t) {
        const _Float16* __restrict__ sprev = sbuf + (size_t)(t & 1) * NB * NN;
        _Float16* __restrict__       snew  = sbuf + (size_t)((t + 1) & 1) * NB * NN;

        // Prefetch this step's external drive BEFORE the GEMM so its latency
        // hides under the 64-WMMA chain (uniform branch, one load clause).
        float xin[8];
        if (observed) {
            const float* __restrict__ sp = spike_input + (size_t)t * NB * NOBS + n_global;
            #pragma unroll
            for (int i = 0; i < 8; ++i)
                xin[i] = sp[(size_t)(i + 8 * h) * NOBS];
        } else {
            #pragma unroll
            for (int i = 0; i < 8; ++i) xin[i] = 0.0f;
        }

        // membrane leak: u *= DECAY
        #pragma unroll
        for (int i = 0; i < 8; ++i) acc[i] *= DECAY;

        // u += s_prev @ W^T (tile): 64 chained v_wmma_f32_16x16x32_f16
        const _Float16* __restrict__ arow = sprev + (size_t)mn * NN;  // A row = batch
        #pragma unroll 8
        for (int k0 = 0; k0 < NN; k0 += 32) {
            const int ka = k0 + 8 * h;
            v8h a0 = *(const v8h*)(arow + ka);        // K = ka..ka+7
            v8h a1 = *(const v8h*)(arow + ka + 16);   // K = ka+16..ka+23
            v8h b0 = *(const v8h*)(wrow + ka);
            v8h b1 = *(const v8h*)(wrow + ka + 16);
            v16h A = __builtin_shufflevector(a0, a1, 0,1,2,3,4,5,6,7,8,9,10,11,12,13,14,15);
            v16h B = __builtin_shufflevector(b0, b1, 0,1,2,3,4,5,6,7,8,9,10,11,12,13,14,15);
            acc = __builtin_amdgcn_wmma_f32_16x16x32_f16(
                      /*neg_a=*/false, A, /*neg_b=*/false, B,
                      /*c_mod=*/(short)0, acc, /*reuse_a=*/false, /*reuse_b=*/false);
        }

        // drive + threshold + emit spikes (C layout: VGPR i, lane l ->
        // m = i + 8*(l>>4), n = l&15). Uniform paths, batched stores.
        float spk[8];
        #pragma unroll
        for (int i = 0; i < 8; ++i) {
            const float u = acc[i] + xin[i] * wscale;
            acc[i] = u;
            spk[i] = (u - THRESH >= 0.0f) ? 1.0f : 0.0f;
        }
        #pragma unroll
        for (int i = 0; i < 8; ++i)
            snew[(size_t)(i + 8 * h) * NN + n_global] = (_Float16)spk[i];
        if (observed) {
            float* __restrict__ op = out + (size_t)t * NB * NOBS + n_global;
            #pragma unroll
            for (int i = 0; i < 8; ++i)
                op[(size_t)(i + 8 * h) * NOBS] = spk[i];
        }

        // device-wide barrier: all spikes of step t visible before step t+1 reads
        __syncthreads();
        if (threadIdx.x == 0) {
            __threadfence();  // release: flush spike stores to device scope
            __hip_atomic_fetch_add(ctr, 1u, __ATOMIC_RELEASE, __HIP_MEMORY_SCOPE_AGENT);
            const unsigned target = (unsigned)(t + 1) * (unsigned)gridDim.x;
            while (__hip_atomic_load(ctr, __ATOMIC_ACQUIRE, __HIP_MEMORY_SCOPE_AGENT) < target)
                __builtin_amdgcn_s_sleep(1);
        }
        __syncthreads();
        __threadfence();      // acquire: invalidate WGP caches so fresh spikes are seen
    }
}

// ---------------------------------------------------------------------
extern "C" void kernel_launch(void* const* d_in, const int* in_sizes, int n_in,
                              void* d_out, int out_size, void* d_ws, size_t ws_size,
                              hipStream_t stream) {
    const float* spike_input  = (const float*)d_in[0];  // [T,B,NOBS]
    const float* W            = (const float*)d_in[1];  // [NN,NN]
    const float* input_weight = (const float*)d_in[2];  // [NOBS]
    float* out = (float*)d_out;                         // [T,B,NOBS]

    char* ws = (char*)d_ws;
    _Float16* Wf16 = (_Float16*)ws;
    _Float16* sbuf = (_Float16*)(ws + W16_BYTES);
    unsigned* ctr  = (unsigned*)(ws + W16_BYTES + SBUF_BYTES);

    rsnn_prep<<<2048, 256, 0, stream>>>(W, Wf16, sbuf, ctr);
    rsnn_step_kernel<<<NUM_WGS, WAVES_PER_WG * 32, 0, stream>>>(
        spike_input, input_weight, Wf16, sbuf, ctr, out);
}
